// RegionalTemporalGCN_31722628448361
// MI455X (gfx1250) — matrix operand order repaired and checked by reading, not compiled, not run on hardware
//
#include <hip/hip_runtime.h>
#include <hip/hip_bf16.h>

#define NN   20000
#define FF   4
#define TT   12
#define CC   256
#define HIDN 128
#define OUTN 12
#define ERE  30000
#define MROWS (NN*TT)          // 240000, divisible by 128
#define LDS_STRIDE 40          // 32 K-halves + 8 pad -> 80B rows, 16B aligned

typedef __attribute__((ext_vector_type(16))) _Float16 v16h;
typedef __attribute__((ext_vector_type(8)))  _Float16 v8h;
typedef __attribute__((ext_vector_type(4)))  _Float16 v4h;
typedef __attribute__((ext_vector_type(8)))  float    v8f;
typedef __attribute__((ext_vector_type(4)))  int      v4i;

#ifndef __has_builtin
#define __has_builtin(x) 0
#endif
#if __has_builtin(__builtin_amdgcn_global_load_async_to_lds_b128) && \
    __has_builtin(__builtin_amdgcn_s_wait_asynccnt)
#define USE_ASYNC_LDS 1
#else
#define USE_ASYNC_LDS 0
#endif

__device__ __forceinline__ void stage_b128(const _Float16* g, _Float16* l) {
#if USE_ASYNC_LDS
    __builtin_amdgcn_global_load_async_to_lds_b128(
        (__attribute__((address_space(1))) v4i*)g,
        (__attribute__((address_space(3))) v4i*)l, 0, 0);
#else
    *(v8h*)l = *(const v8h*)g;
#endif
}

__device__ __forceinline__ float fast_sigmoid(float x) {
    // exp2/rcp saturate correctly at the extremes, no range fixup needed
    return __builtin_amdgcn_rcpf(1.0f + __builtin_amdgcn_exp2f(-1.44269504f * x));
}
__device__ __forceinline__ float fast_tanh(float x) {
#if __has_builtin(__builtin_amdgcn_tanhf)
    return __builtin_amdgcn_tanhf(x);   // V_TANH_F32
#else
    return tanhf(x);
#endif
}

// ---------------- degree / normalization ----------------
__global__ void deg_init_kernel(float* deg) {
    int i = blockIdx.x * 256 + threadIdx.x;
    if (i < 5 * NN) deg[i] = 1.0f;   // self-loop contributes 1
}

__global__ void deg_acc_kernel(const int* __restrict__ dst,
                               const float* __restrict__ w,
                               float* __restrict__ deg_r) {
    int e = blockIdx.x * 256 + threadIdx.x;
    if (e < ERE) atomicAdd(&deg_r[dst[e]], w[e]);
}

__global__ void deg_fin_kernel(float* __restrict__ dinv, float* __restrict__ dsum) {
    int n = blockIdx.x * 256 + threadIdx.x;
    if (n >= NN) return;
    float s = 0.f;
    for (int r = 0; r < 5; ++r) {
        float di = rsqrtf(dinv[r * NN + n]);
        dinv[r * NN + n] = di;
        s += di * di;
    }
    dsum[n] = s;
}

// ---------------- acc init: bc + (sum_r dinv^2) * xw (xw recomputed, F=4) ----
__global__ void acc_init_kernel(const float* __restrict__ x,
                                const float* __restrict__ dsum,
                                const float* __restrict__ WcZ, const float* __restrict__ bcZ,
                                const float* __restrict__ WcH, const float* __restrict__ bcH,
                                float* __restrict__ accZ, float* __restrict__ accH) {
    size_t idx = (size_t)blockIdx.x * 256 + threadIdx.x;   // over N*T*C
    if (idx >= (size_t)NN * TT * CC) return;
    int c = (int)(idx & (CC - 1));
    int t = (int)((idx >> 8) % TT);
    int n = (int)(idx / (TT * CC));
    const float* xs = x + (size_t)n * (FF * TT);
    float x0 = xs[0 * TT + t], x1 = xs[1 * TT + t], x2 = xs[2 * TT + t], x3 = xs[3 * TT + t];
    float xwz = x0 * WcZ[c] + x1 * WcZ[CC + c] + x2 * WcZ[2 * CC + c] + x3 * WcZ[3 * CC + c];
    float xwh = x0 * WcH[c] + x1 * WcH[CC + c] + x2 * WcH[2 * CC + c] + x3 * WcH[3 * CC + c];
    float ds = dsum[n];
    accZ[idx] = bcZ[c] + ds * xwz;
    accH[idx] = bcH[c] + ds * xwh;
}

// ---------------- edge scatter: one block per edge, both gates ---------------
__global__ __launch_bounds__(256) void scatter_kernel(
    const int* __restrict__ src, const int* __restrict__ dst,
    const float* __restrict__ w, const float* __restrict__ dinv_r,
    const float* __restrict__ x,
    const float* __restrict__ WcZ, const float* __restrict__ WcH,
    float* __restrict__ accZ, float* __restrict__ accH) {
    int e = blockIdx.x;
    int s = src[e], d = dst[e];
    float coef = dinv_r[s] * w[e] * dinv_r[d];
    const float* xs = x + (size_t)s * (FF * TT);
    size_t dbase = (size_t)d * (TT * CC);
    for (int tc = threadIdx.x; tc < TT * CC; tc += 256) {
        int t = tc >> 8, c = tc & (CC - 1);
        float x0 = xs[0 * TT + t], x1 = xs[1 * TT + t], x2 = xs[2 * TT + t], x3 = xs[3 * TT + t];
        float xwz = x0 * WcZ[c] + x1 * WcZ[CC + c] + x2 * WcZ[2 * CC + c] + x3 * WcZ[3 * CC + c];
        float xwh = x0 * WcH[c] + x1 * WcH[CC + c] + x2 * WcH[2 * CC + c] + x3 * WcH[3 * CC + c];
        atomicAdd(&accZ[dbase + tc], coef * xwz);
        atomicAdd(&accH[dbase + tc], coef * xwh);
    }
}

// ---------------- weight conversion to f16, TRANSPOSED: wlT[co][ci] ---------
__global__ void wl_cvt_kernel(const float* __restrict__ WlZ, const float* __restrict__ WlH,
                              _Float16* __restrict__ wlz16T, _Float16* __restrict__ wlh16T) {
    int i = blockIdx.x * 256 + threadIdx.x;   // i = co*CC + ci
    if (i < CC * CC) {
        int co = i >> 8, ci = i & (CC - 1);
        wlz16T[i] = (_Float16)WlZ[ci * CC + co];
        wlh16T[i] = (_Float16)WlH[ci * CC + co];
    }
}

// ---------------- acc f32 -> f16 operand conversion -------------------------
__global__ void acc_cvt_kernel(const float* __restrict__ accZ, const float* __restrict__ accH,
                               _Float16* __restrict__ a16Z, _Float16* __restrict__ a16H) {
    size_t i4 = ((size_t)blockIdx.x * 256 + threadIdx.x) * 4;
    if (i4 >= (size_t)MROWS * CC) return;
    float4 z = *(const float4*)(accZ + i4);
    float4 h = *(const float4*)(accH + i4);
    v4h zz = { (_Float16)z.x, (_Float16)z.y, (_Float16)z.z, (_Float16)z.w };
    v4h hh = { (_Float16)h.x, (_Float16)h.y, (_Float16)h.z, (_Float16)h.w };
    *(v4h*)(a16Z + i4) = zz;
    *(v4h*)(a16H + i4) = hh;
}

// ---------------- fragment loaders (ISA 7.12.2 layouts) ---------------------
__device__ __forceinline__ v16h load_fragA(const _Float16* base, int lane) {
    // lanes 0-15: row=lane, K={0..7, 16..23}; lanes 16-31: row=lane-16, K={8..15, 24..31}
    const _Float16* p = base + (lane & 15) * LDS_STRIDE + ((lane & 16) ? 8 : 0);
    v8h lo = *(const v8h*)(p);        // K = kb .. kb+7
    v8h hi = *(const v8h*)(p + 16);   // K = kb+16 .. kb+23
    return __builtin_shufflevector(lo, hi, 0,1,2,3,4,5,6,7,8,9,10,11,12,13,14,15);
}
__device__ __forceinline__ v16h load_fragB(const _Float16* base, int lane) {
    // lanes 0-15: col=lane, K={0..7, 16..23}; lanes 16-31: col=lane-16, K={8..15, 24..31}
    const _Float16* p = base + (lane & 15) * LDS_STRIDE + ((lane & 16) ? 8 : 0);
    v8h lo = *(const v8h*)(p);
    v8h hi = *(const v8h*)(p + 16);
    return __builtin_shufflevector(lo, hi, 0,1,2,3,4,5,6,7,8,9,10,11,12,13,14,15);
}

// ---------------- fused dual-gate GEMM + GRU gate epilogue ------------------
// Ht[m, co] = (1 - sigmoid(A_z[m,:]@WlZ + blZ)) * tanh(A_h[m,:]@WlH + blH)
__global__ __launch_bounds__(256) void gemm_gate_kernel(
    const _Float16* __restrict__ a16Z, const _Float16* __restrict__ a16H,
    const _Float16* __restrict__ wlz16T, const _Float16* __restrict__ wlh16T,
    const float* __restrict__ blZ, const float* __restrict__ blH,
    float* __restrict__ Ht) {
    __shared__ __align__(16) _Float16 sAz[128 * LDS_STRIDE];
    __shared__ __align__(16) _Float16 sAh[128 * LDS_STRIDE];
    __shared__ __align__(16) _Float16 sBz[64 * LDS_STRIDE];   // [col][k]
    __shared__ __align__(16) _Float16 sBh[64 * LDS_STRIDE];

    const int tid  = threadIdx.x;
    const int lane = tid & 31;
    const int wave = tid >> 5;
    const int wm = wave & 3;                 // 4 waves along M
    const int wn = wave >> 2;                // 2 waves along N
    const int row0 = blockIdx.x * 128;       // 1875 blocks, exact
    const int col0 = blockIdx.y * 64;        // 4 blocks, exact

    v8f cz[2][2] = {};
    v8f ch[2][2] = {};

    for (int k0 = 0; k0 < CC; k0 += 32) {
        // stage A tiles: 128 rows x 64B = 512 chunks of 16B per gate
        for (int i = 0; i < 2; ++i) {
            int chunk = i * 256 + tid;
            int r = chunk >> 2, q = chunk & 3;
            size_t goff = (size_t)(row0 + r) * CC + k0 + q * 8;
            stage_b128(a16Z + goff, &sAz[r * LDS_STRIDE + q * 8]);
            stage_b128(a16H + goff, &sAh[r * LDS_STRIDE + q * 8]);
        }
        // stage B tiles (pre-transposed): 64 rows x 64B = 256 chunks per gate
        {
            int r = tid >> 2, q = tid & 3;
            size_t goff = (size_t)(col0 + r) * CC + k0 + q * 8;
            stage_b128(wlz16T + goff, &sBz[r * LDS_STRIDE + q * 8]);
            stage_b128(wlh16T + goff, &sBh[r * LDS_STRIDE + q * 8]);
        }
#if USE_ASYNC_LDS
        __builtin_amdgcn_s_wait_asynccnt(0);
#endif
        __syncthreads();
        v16h az[2], ah[2], bz[2], bh[2];
        for (int mi = 0; mi < 2; ++mi) {
            az[mi] = load_fragA(&sAz[(wm * 32 + mi * 16) * LDS_STRIDE], lane);
            ah[mi] = load_fragA(&sAh[(wm * 32 + mi * 16) * LDS_STRIDE], lane);
        }
        for (int ni = 0; ni < 2; ++ni) {
            bz[ni] = load_fragB(&sBz[(wn * 32 + ni * 16) * LDS_STRIDE], lane);
            bh[ni] = load_fragB(&sBh[(wn * 32 + ni * 16) * LDS_STRIDE], lane);
        }
        for (int mi = 0; mi < 2; ++mi)
            for (int ni = 0; ni < 2; ++ni) {
                cz[mi][ni] = __builtin_amdgcn_wmma_f32_16x16x32_f16(
                    false, az[mi], false, bz[ni], (short)0, cz[mi][ni], false, false);
                ch[mi][ni] = __builtin_amdgcn_wmma_f32_16x16x32_f16(
                    false, ah[mi], false, bh[ni], (short)0, ch[mi][ni], false, false);
            }
        __syncthreads();
    }
    // epilogue: gate math (straight-line TRANS ops), write Ht
    for (int mi = 0; mi < 2; ++mi)
        for (int ni = 0; ni < 2; ++ni) {
            int colL = wn * 32 + ni * 16 + (lane & 15);
            float bZ = blZ[col0 + colL];
            float bH = blH[col0 + colL];
            for (int j = 0; j < 8; ++j) {
                int rowL = wm * 32 + mi * 16 + ((lane & 16) ? (j + 8) : j);
                float sz = fast_sigmoid(cz[mi][ni][j] + bZ);
                float th = fast_tanh(ch[mi][ni][j] + bH);
                Ht[(size_t)(row0 + rowL) * CC + (col0 + colL)] = (1.0f - sz) * th;
            }
        }
}

// ---------------- attention softmax (T=12) ----------------------------------
__global__ void probs_kernel(const float* __restrict__ att, float* __restrict__ probs) {
    if (threadIdx.x == 0 && blockIdx.x == 0) {
        float m = att[0];
        for (int t = 1; t < TT; ++t) m = fmaxf(m, att[t]);
        float e[TT]; float s = 0.f;
        for (int t = 0; t < TT; ++t) { e[t] = __expf(att[t] - m); s += e[t]; }
        float inv = 1.0f / s;
        for (int t = 0; t < TT; ++t) probs[t] = e[t] * inv;
    }
}

// ---------------- temporal pooling: h[n,c] = sum_t probs[t]*Ht ---------------
__global__ void pool_kernel(const float* __restrict__ Ht, const float* __restrict__ probs,
                            float* __restrict__ hout) {
    size_t idx = (size_t)blockIdx.x * 256 + threadIdx.x;
    if (idx >= (size_t)NN * CC) return;
    int n = (int)(idx / CC), c = (int)(idx % CC);
    float s = 0.f;
    for (int t = 0; t < TT; ++t)
        s += probs[t] * Ht[((size_t)n * TT + t) * CC + c];
    hout[idx] = s;
}

// ---------------- MLP head ---------------------------------------------------
__global__ void mlp1_kernel(const float* __restrict__ h, const float* __restrict__ W1,
                            const float* __restrict__ b1, float* __restrict__ hid) {
    size_t idx = (size_t)blockIdx.x * 256 + threadIdx.x;
    if (idx >= (size_t)NN * HIDN) return;
    int n = (int)(idx / HIDN), j = (int)(idx % HIDN);
    const float* hr = h + (size_t)n * CC;
    float s = b1[j];
    for (int c = 0; c < CC; ++c) s += fmaxf(hr[c], 0.f) * W1[c * HIDN + j];
    hid[idx] = fmaxf(s, 0.f);
}

__global__ void mlp2_kernel(const float* __restrict__ hid, const float* __restrict__ W2,
                            const float* __restrict__ b2, float* __restrict__ y) {
    int idx = blockIdx.x * 256 + threadIdx.x;
    if (idx >= NN * OUTN) return;
    int n = idx / OUTN, o = idx % OUTN;
    float s = b2[o];
    const float* hr = hid + (size_t)n * HIDN;
    for (int j = 0; j < HIDN; ++j) s += hr[j] * W2[j * OUTN + o];
    y[idx] = s;
}

// ---------------- host orchestration ----------------------------------------
extern "C" void kernel_launch(void* const* d_in, const int* in_sizes, int n_in,
                              void* d_out, int out_size, void* d_ws, size_t ws_size,
                              hipStream_t stream) {
    const float* x   = (const float*)d_in[0];
    // d_in[1] = full edge_index (unused by reference output)
    const int*   ei[5]  = { (const int*)d_in[2], (const int*)d_in[3], (const int*)d_in[4],
                            (const int*)d_in[5], (const int*)d_in[6] };
    const float* ew[5]  = { (const float*)d_in[7], (const float*)d_in[8], (const float*)d_in[9],
                            (const float*)d_in[10], (const float*)d_in[11] };
    const float* WcZ = (const float*)d_in[12];
    const float* bcZ = (const float*)d_in[13];
    const float* WlZ = (const float*)d_in[14];
    const float* blZ = (const float*)d_in[15];
    // 16..19: r-gate (cannot affect output)
    const float* WcH = (const float*)d_in[20];
    const float* bcH = (const float*)d_in[21];
    const float* WlH = (const float*)d_in[22];
    const float* blH = (const float*)d_in[23];
    const float* att = (const float*)d_in[24];
    const float* W1  = (const float*)d_in[25];
    const float* b1  = (const float*)d_in[26];
    const float* W2  = (const float*)d_in[27];
    const float* b2  = (const float*)d_in[28];

    float* y_out = (float*)d_out;                       // [N, OUT]
    float* h_out = (float*)d_out + (size_t)NN * OUTN;   // [N, C]

    // workspace layout
    char* ws = (char*)d_ws;
    size_t off = 0;
    auto alloc = [&](size_t bytes) { char* p = ws + off; off = (off + bytes + 255) & ~(size_t)255; return p; };
    float*    dinv   = (float*)alloc((size_t)5 * NN * sizeof(float));   // deg -> dinv in-place
    float*    dsum   = (float*)alloc((size_t)NN * sizeof(float));
    float*    probs  = (float*)alloc(TT * sizeof(float));
    _Float16* wlz16T = (_Float16*)alloc((size_t)CC * CC * sizeof(_Float16));
    _Float16* wlh16T = (_Float16*)alloc((size_t)CC * CC * sizeof(_Float16));
    float*    accZ   = (float*)alloc((size_t)MROWS * CC * sizeof(float));
    float*    accH   = (float*)alloc((size_t)MROWS * CC * sizeof(float));
    _Float16* a16Z   = (_Float16*)alloc((size_t)MROWS * CC * sizeof(_Float16));
    _Float16* a16H   = (_Float16*)alloc((size_t)MROWS * CC * sizeof(_Float16));
    float*    Htb    = accZ;   // reuse: GEMM reads only a16Z/a16H (accZ free after cvt)
    float*    hid    = accH;   // reuse: free after cvt as well

    // 1. degrees
    deg_init_kernel<<<(5 * NN + 255) / 256, 256, 0, stream>>>(dinv);
    for (int r = 0; r < 5; ++r)
        deg_acc_kernel<<<(ERE + 255) / 256, 256, 0, stream>>>(ei[r] + ERE, ew[r], dinv + r * NN);
    deg_fin_kernel<<<(NN + 255) / 256, 256, 0, stream>>>(dinv, dsum);

    // 2. weight conversion (transposed) + attention softmax (independent)
    wl_cvt_kernel<<<(CC * CC + 255) / 256, 256, 0, stream>>>(WlZ, WlH, wlz16T, wlh16T);
    probs_kernel<<<1, 32, 0, stream>>>(att, probs);

    // 3. accumulator init (bias + self-loop term, xw recomputed on the fly)
    {
        size_t total = (size_t)NN * TT * CC;
        acc_init_kernel<<<(unsigned)((total + 255) / 256), 256, 0, stream>>>(
            x, dsum, WcZ, bcZ, WcH, bcH, accZ, accH);
    }

    // 4. edge scatter-add per region (one block per edge)
    for (int r = 0; r < 5; ++r)
        scatter_kernel<<<ERE, 256, 0, stream>>>(ei[r], ei[r] + ERE, ew[r], dinv + r * NN,
                                                x, WcZ, WcH, accZ, accH);

    // 5. convert accumulators to f16 WMMA operands
    {
        size_t q = (size_t)MROWS * CC / 4;
        acc_cvt_kernel<<<(unsigned)((q + 255) / 256), 256, 0, stream>>>(accZ, accH, a16Z, a16H);
    }

    // 6. fused dual GEMM + GRU gating -> Ht (overwrites accZ region)
    gemm_gate_kernel<<<dim3(MROWS / 128, CC / 64), 256, 0, stream>>>(
        a16Z, a16H, wlz16T, wlh16T, blZ, blH, Htb);

    // 7. attention pooling -> h (second output)
    pool_kernel<<<(unsigned)(((size_t)NN * CC + 255) / 256), 256, 0, stream>>>(Htb, probs, h_out);

    // 8. MLP head -> y (first output)
    mlp1_kernel<<<(unsigned)(((size_t)NN * HIDN + 255) / 256), 256, 0, stream>>>(h_out, W1, b1, hid);
    mlp2_kernel<<<(NN * OUTN + 255) / 256, 256, 0, stream>>>(hid, W2, b2, y_out);
}
